// RNN_LSTM_43121471652257
// MI455X (gfx1250) — compile-verified, matrix operand order checked by reading, not run on hardware
//
#include <hip/hip_runtime.h>
#include <hip/hip_bf16.h>

#if defined(__has_builtin)
#if __has_builtin(__builtin_amdgcn_global_load_async_to_lds_b128)
#define USE_ASYNC_LDS 1
#endif
#endif

// ---------- types ----------
typedef __bf16 bf16;
typedef __attribute__((ext_vector_type(16))) __bf16 v16bf;
typedef __attribute__((ext_vector_type(8)))  float  v8f;
typedef int vi4_ __attribute__((vector_size(4 * sizeof(int))));  // b128 payload
typedef __attribute__((address_space(1))) vi4_ gvi4;             // global
typedef __attribute__((address_space(3))) vi4_ lvi4;             // LDS

union Frag16 { v16bf v; unsigned int u[8]; };
union Gate4  { unsigned long long q; bf16 h[4]; };

// K index (start of the even pair) inside a 16x32 bf16 A/B fragment.
// ISA 7.12.2: VGPR v holds a K pair; lanes 0-15 vs 16-31 offset K by 8.
__device__ __forceinline__ int kidx(int v, int g) {
    return ((v & 4) ? 16 : 0) + ((v & 3) << 1) + (g << 3);
}

// Load one 16x32 bf16 fragment. p0 -> element (row 0, k 0) of the tile,
// ld = row stride in elements (must be even). Works for LDS or global.
__device__ __forceinline__ v16bf frag16(const bf16* p0, int ld, int lane) {
    Frag16 f;
    const int g = lane >> 4;
    const bf16* p = p0 + (size_t)(lane & 15) * ld;
#pragma unroll
    for (int v = 0; v < 8; ++v)
        f.u[v] = *(const unsigned int*)(p + kidx(v, g));
    return f.v;
}

__device__ __forceinline__ v8f wmma_bf16(v16bf a, v16bf b, v8f c) {
    // (neg_a, A, neg_b, B, c_mod, C, reuse_a, reuse_b)
    return __builtin_amdgcn_wmma_f32_16x16x32_bf16(false, a, false, b, (short)0, c, false, false);
}

__device__ __forceinline__ float sigmoidf_(float x) {
    return 1.0f / (1.0f + __expf(-x));
}

// ---------- small prep kernels ----------
__global__ void cast_f32_bf16_kernel(const float* __restrict__ src,
                                     bf16* __restrict__ dst, int n) {
    int i = blockIdx.x * blockDim.x + threadIdx.x;
    if (i < n) dst[i] = (bf16)src[i];
}

__global__ void bias_sum_kernel(const float* __restrict__ a,
                                const float* __restrict__ b,
                                float* __restrict__ d, int n) {
    int i = blockIdx.x * blockDim.x + threadIdx.x;
    if (i < n) d[i] = a[i] + b[i];
}

// emb gather: x0[(s*B+b)*128 + e] = bf16(emb[tok*128+e]); inputs is [B,S]
__global__ void gather_kernel(const int* __restrict__ inputs,
                              const float* __restrict__ emb,
                              bf16* __restrict__ x0) {
    int i = blockIdx.x * blockDim.x + threadIdx.x;   // S*B*128 threads
    int e = i & 127;
    int m = i >> 7;          // m = s*B + b
    int s = m >> 7;          // B == 128
    int b = m & 127;
    int tok = inputs[b * 1024 + s];
    x0[(size_t)m * 128 + e] = (bf16)emb[(size_t)tok * 128 + e];
}

// ---------- WMMA GEMM producing gate-interleaved activations ----------
// G[m, dir*512 + h*4 + g] = sum_k A[m,k]*W[n,k] + bias[n],  n = dir*512+g*128+h
// block tile 128x128, 8 waves, each wave 64x32 (4 M-subtiles x 2 N-subtiles)
__global__ __launch_bounds__(256, 2)
void gemm_bias_gates(const bf16* __restrict__ A, const bf16* __restrict__ W,
                     const float* __restrict__ bias, bf16* __restrict__ G,
                     int M, int N, int K) {
    const int LDK = 40;                  // padded K-panel stride (elements)
    __shared__ bf16 As[128 * 40];
    __shared__ bf16 Ws[128 * 40];

    const int tid  = threadIdx.x;
    const int lane = tid & 31;
    const int w    = tid >> 5;
    const int wm   = w >> 2;             // 0..1 -> 64 rows
    const int wn   = w & 3;              // 0..3 -> 32 cols
    const int l16  = lane & 15;
    const int g16  = lane >> 4;
    const size_t mbase = (size_t)blockIdx.y * 128;
    const int    nbase = blockIdx.x * 128;

    v8f acc[4][2] = {};

    for (int k0 = 0; k0 < K; k0 += 32) {
        // cooperative panel load: each thread moves 16 halfs (32 B) of A and W
        int r    = tid >> 1;
        int half = (tid & 1) * 16;
        const bf16* ag = A + (mbase + r) * (size_t)K + k0 + half;
        const bf16* wg = W + (size_t)(nbase + r) * K + k0 + half;
#ifdef USE_ASYNC_LDS
        // async global->LDS copy, tracked by ASYNCcnt (no VGPR staging)
        __builtin_amdgcn_global_load_async_to_lds_b128(
            (gvi4*)ag,       (lvi4*)(As + r * LDK + half),     0, 0);
        __builtin_amdgcn_global_load_async_to_lds_b128(
            (gvi4*)(ag + 8), (lvi4*)(As + r * LDK + half + 8), 0, 0);
        __builtin_amdgcn_global_load_async_to_lds_b128(
            (gvi4*)wg,       (lvi4*)(Ws + r * LDK + half),     0, 0);
        __builtin_amdgcn_global_load_async_to_lds_b128(
            (gvi4*)(wg + 8), (lvi4*)(Ws + r * LDK + half + 8), 0, 0);
#if __has_builtin(__builtin_amdgcn_s_wait_asynccnt)
        __builtin_amdgcn_s_wait_asynccnt(0);
#else
        asm volatile("s_wait_asynccnt 0x0" ::: "memory");
#endif
#else
        uint4 a0 = ((const uint4*)ag)[0];
        uint4 a1 = ((const uint4*)ag)[1];
        uint4 w0 = ((const uint4*)wg)[0];
        uint4 w1 = ((const uint4*)wg)[1];
        *(uint4*)(As + r * LDK + half)     = a0;
        *(uint4*)(As + r * LDK + half + 8) = a1;
        *(uint4*)(Ws + r * LDK + half)     = w0;
        *(uint4*)(Ws + r * LDK + half + 8) = w1;
#endif
        __syncthreads();

        v16bf af[4], bfv[2];
#pragma unroll
        for (int mt = 0; mt < 4; ++mt)
            af[mt] = frag16(As + (wm * 64 + mt * 16) * LDK, LDK, lane);
#pragma unroll
        for (int nt = 0; nt < 2; ++nt)
            bfv[nt] = frag16(Ws + (wn * 32 + nt * 16) * LDK, LDK, lane);
#pragma unroll
        for (int mt = 0; mt < 4; ++mt)
#pragma unroll
            for (int nt = 0; nt < 2; ++nt)
                acc[mt][nt] = wmma_bf16(af[mt], bfv[nt], acc[mt][nt]);
        __syncthreads();
    }

#pragma unroll
    for (int nt = 0; nt < 2; ++nt) {
        const int n   = nbase + wn * 32 + nt * 16 + l16;  // gate-major column
        const int dir = n >> 9;
        const int gat = (n >> 7) & 3;
        const int hh  = n & 127;
        const int col = dir * 512 + hh * 4 + gat;          // interleaved column
        const float bv = bias[n];
#pragma unroll
        for (int mt = 0; mt < 4; ++mt) {
#pragma unroll
            for (int r = 0; r < 8; ++r) {
                size_t m = mbase + wm * 64 + mt * 16 + r + 8 * g16;
                G[m * (size_t)N + col] = (bf16)(acc[mt][nt][r] + bv);
            }
        }
    }
}

// ---------- LSTM scan ----------
// grid: (8 batch-slices of 16 rows, 2 directions), 256 threads (8 waves).
// xg: [S*B, 1024] pre-biased gates, interleaved: col = dir*512 + h*4 + gate
// whh: [2, 512, 128] bf16. Whh fragments are preloaded into VGPRs once.
// Wave w owns hidden columns [w*16, w*16+16); c lives in registers.
__global__ __launch_bounds__(256, 1)
void lstm_scan(const bf16* __restrict__ xg, const bf16* __restrict__ whh,
               bf16* __restrict__ hs_out, float* __restrict__ hfin) {
    const int S_ = 1024, B_ = 128;
    const int LDH = 132;                       // padded h row stride (bank-safe)
    __shared__ bf16 h_sh[16 * 132];

    const int tid  = threadIdx.x;
    const int lane = tid & 31;
    const int w    = tid >> 5;
    const int l16  = lane & 15;
    const int g16  = lane >> 4;
    const int dir   = blockIdx.y;
    const int bbase = blockIdx.x * 16;
    const bf16* whhD = whh + (size_t)dir * 512 * 128;
    const int hcol  = w * 16 + l16;            // this lane's hidden column
    const size_t xoff = (size_t)dir * 512 + (size_t)hcol * 4;

    // preload B fragments of Whh^T: B(k,n) = whh[n*128+k]; n = gate*128+hcol
    v16bf bfr[4][4];
#pragma unroll
    for (int gg = 0; gg < 4; ++gg)
#pragma unroll
        for (int kb = 0; kb < 4; ++kb)
            bfr[gg][kb] = frag16(whhD + (size_t)(gg * 128 + w * 16) * 128 + kb * 32,
                                 128, lane);

    float cc[8];
#pragma unroll
    for (int r = 0; r < 8; ++r) cc[r] = 0.0f;

    for (int i = tid; i < 16 * LDH; i += 256) h_sh[i] = (bf16)0.0f;
    __syncthreads();

    for (int t = 0; t < S_; ++t) {
        const int s = dir ? (S_ - 1 - t) : t;
        const size_t mrow = (size_t)s * B_ + bbase;

        // read h fragments (A: 16 batch rows x 128 K)
        v16bf af[4];
#pragma unroll
        for (int kb = 0; kb < 4; ++kb)
            af[kb] = frag16(h_sh + kb * 32, LDH, lane);
        __syncthreads();                       // all reads done before writes

        v8f acc[4];
#pragma unroll
        for (int gg = 0; gg < 4; ++gg) {
            v8f a = {};
#pragma unroll
            for (int kb = 0; kb < 4; ++kb)
                a = wmma_bf16(af[kb], bfr[gg][kb], a);
            acc[gg] = a;
        }

        bf16 hbv[8];
#pragma unroll
        for (int r = 0; r < 8; ++r) {
            const size_t m = mrow + r + 8 * g16;
            Gate4 gq;                           // one 8B load = all 4 gates
            gq.q = *(const unsigned long long*)(xg + m * 1024 + xoff);
            float iv = acc[0][r] + (float)gq.h[0];
            float fv = acc[1][r] + (float)gq.h[1];
            float gv = acc[2][r] + (float)gq.h[2];
            float ov = acc[3][r] + (float)gq.h[3];
            float c  = sigmoidf_(fv) * cc[r] + sigmoidf_(iv) * tanhf(gv);
            cc[r] = c;
            hbv[r] = (bf16)(sigmoidf_(ov) * tanhf(c));
        }

        // prefetch next step's gate row into L2/L1
        if (t + 1 < S_) {
            const int sn = dir ? (s - 1) : (s + 1);
            __builtin_prefetch(xg + ((size_t)sn * B_ + bbase + 8 * g16) * 1024 + xoff,
                               0, 1);
        }

        // publish h for next step (+ full sequence output for layer 0)
#pragma unroll
        for (int r = 0; r < 8; ++r) {
            const int row = r + 8 * g16;
            h_sh[row * LDH + hcol] = hbv[r];
            if (hs_out)
                hs_out[(mrow + row) * 256 + dir * 128 + hcol] = hbv[r];
        }
        __syncthreads();
    }

    if (hfin) {
        for (int i = tid; i < 16 * 128; i += 256) {
            const int row = i >> 7, col = i & 127;
            hfin[(size_t)(bbase + row) * 256 + dir * 128 + col] =
                (float)h_sh[row * LDH + col];
        }
    }
}

// ---------- tiny f32 decoder: out = (h @ decW^T + decb) @ fcW^T + fcb ----------
__global__ void decode_kernel(const float* __restrict__ hf,
                              const float* __restrict__ decW,
                              const float* __restrict__ decb,
                              const float* __restrict__ fcW,
                              const float* __restrict__ fcb,
                              float* __restrict__ out) {
    const int b = threadIdx.x;                 // 128 threads, one batch row each
    if (b >= 128) return;
    float d[64];
#pragma unroll 4
    for (int j = 0; j < 64; ++j) {
        float acc = decb[j];
        for (int k = 0; k < 256; ++k)
            acc += hf[b * 256 + k] * decW[j * 256 + k];
        d[j] = acc;
    }
    for (int c = 0; c < 2; ++c) {
        float acc = fcb[c];
#pragma unroll
        for (int j = 0; j < 64; ++j) acc += d[j] * fcW[c * 64 + j];
        out[b * 2 + c] = acc;
    }
}

// ---------- launch ----------
extern "C" void kernel_launch(void* const* d_in, const int* in_sizes, int n_in,
                              void* d_out, int out_size, void* d_ws, size_t ws_size,
                              hipStream_t stream) {
    const int*   inputs = (const int*)  d_in[0];   // [128,1024]
    const float* emb    = (const float*)d_in[1];   // [50000,128]
    const float* Wih0   = (const float*)d_in[2];   // [2,512,128]
    const float* Whh0   = (const float*)d_in[3];   // [2,512,128]
    const float* bih0   = (const float*)d_in[4];   // [2,512]
    const float* bhh0   = (const float*)d_in[5];
    const float* Wih1   = (const float*)d_in[6];   // [2,512,256]
    const float* Whh1   = (const float*)d_in[7];   // [2,512,128]
    const float* bih1   = (const float*)d_in[8];
    const float* bhh1   = (const float*)d_in[9];
    const float* decW   = (const float*)d_in[10];  // [64,256]
    const float* decb   = (const float*)d_in[11];
    const float* fc1W   = (const float*)d_in[12];  // [2,64]
    const float* fc1b   = (const float*)d_in[13];
    float* out = (float*)d_out;                    // [128,2]

    char* wsb = (char*)d_ws;
    auto alloc = [&](size_t bytes) -> char* {
        char* p = wsb;
        wsb += (bytes + 255) & ~(size_t)255;
        return p;
    };
    const size_t M = (size_t)1024 * 128;           // S*B = 131072
    bf16*  W0c = (bf16*)alloc(1024 * 128 * 2);     // Wih_l0 flat -> [1024,128]
    bf16*  W1c = (bf16*)alloc(1024 * 256 * 2);     // Wih_l1 flat -> [1024,256]
    bf16*  Wh0 = (bf16*)alloc(2 * 512 * 128 * 2);
    bf16*  Wh1 = (bf16*)alloc(2 * 512 * 128 * 2);
    float* b0  = (float*)alloc(1024 * 4);
    float* b1  = (float*)alloc(1024 * 4);
    bf16*  X0  = (bf16*)alloc(M * 128 * 2);        // 32 MB
    bf16*  HS  = (bf16*)alloc(M * 256 * 2);        // 64 MB
    bf16*  XG  = (bf16*)alloc(M * 1024 * 2);       // 256 MB (reused both layers)
    float* HF  = (float*)alloc(128 * 256 * 4);

    cast_f32_bf16_kernel<<<(1024 * 128 + 255) / 256, 256, 0, stream>>>(Wih0, W0c, 1024 * 128);
    cast_f32_bf16_kernel<<<(1024 * 256 + 255) / 256, 256, 0, stream>>>(Wih1, W1c, 1024 * 256);
    cast_f32_bf16_kernel<<<(2 * 512 * 128 + 255) / 256, 256, 0, stream>>>(Whh0, Wh0, 2 * 512 * 128);
    cast_f32_bf16_kernel<<<(2 * 512 * 128 + 255) / 256, 256, 0, stream>>>(Whh1, Wh1, 2 * 512 * 128);
    bias_sum_kernel<<<4, 256, 0, stream>>>(bih0, bhh0, b0, 1024);
    bias_sum_kernel<<<4, 256, 0, stream>>>(bih1, bhh1, b1, 1024);

    gather_kernel<<<(int)(M * 128 / 256), 256, 0, stream>>>(inputs, emb, X0);

    // layer 0: input projection (K=128) then scan (writes full sequence HS)
    gemm_bias_gates<<<dim3(8, 1024), 256, 0, stream>>>(X0, W0c, b0, XG,
                                                       (int)M, 1024, 128);
    lstm_scan<<<dim3(8, 2), 256, 0, stream>>>(XG, Wh0, HS, nullptr);

    // layer 1: input projection (K=256) then scan (writes only final h)
    gemm_bias_gates<<<dim3(8, 1024), 256, 0, stream>>>(HS, W1c, b1, XG,
                                                       (int)M, 1024, 256);
    lstm_scan<<<dim3(8, 2), 256, 0, stream>>>(XG, Wh1, nullptr, HF);

    decode_kernel<<<1, 128, 0, stream>>>(HF, decW, decb, fc1W, fc1b, out);
}
